// AbLang2MHA_80496277062506
// MI455X (gfx1250) — matrix operand order repaired and checked by reading, not compiled, not run on hardware
//
#include <hip/hip_runtime.h>

// ---------------------------------------------------------------------------
// AbLang2 MHA block on MI455X (gfx1250): bf16 WMMA GEMMs + flash attention.
// All matrix math via v_wmma_f32_16x16x32_bf16 (f32 accumulate).
// GEMM wave tile: 64x64 (4x4 WMMA grid) -> 16 WMMAs per 16 b128 loads.
// ---------------------------------------------------------------------------

#define BATCH   2
#define SEQ     2048
#define NHEADS  16
#define HDIM    64
#define DMODEL  1024
#define TOK     (BATCH * SEQ)          // 4096 tokens
#define QSCALE  0.015625f              // SCALING(0.125) / sqrt(HDIM)(8)

typedef __attribute__((ext_vector_type(16))) __bf16 v16bf;
typedef __attribute__((ext_vector_type(8)))  __bf16 v8bf;
typedef __attribute__((ext_vector_type(8)))  float  v8f;

// ---- scalar bf16 helpers (RNE) --------------------------------------------
__device__ __forceinline__ unsigned short f2bf(float f) {
    unsigned int u = __float_as_uint(f);
    unsigned int r = (u + 0x7FFFu + ((u >> 16) & 1u)) >> 16;
    return (unsigned short)r;
}
__device__ __forceinline__ float bf2f(unsigned short h) {
    return __uint_as_float(((unsigned int)h) << 16);
}

// ---- WMMA fragment loads ---------------------------------------------------
// A fragment (16x32 bf16, M x K): lane = row (lane&15); lanes<16 hold
// K = 0..15 contiguous, lanes>=16 hold K = 16..31.  Caller passes a
// pointer to the first 8-element chunk; second chunk is +8 elements.
__device__ __forceinline__ v16bf load_fragA(const unsigned short* p) {
    v8bf lo = *(const v8bf*)p;
    v8bf hi = *(const v8bf*)(p + 8);
    return __builtin_shufflevector(lo, hi, 0,1,2,3,4,5,6,7,8,9,10,11,12,13,14,15);
}
// B fragment (32x16 bf16, K x N): lane = column (lane&15); lanes<16 hold
// K = 0..15 contiguous, lanes>=16 hold K = 16..31.  Caller passes pointer to
// the 16-element contiguous run for this lane.
__device__ __forceinline__ v16bf load_fragB(const unsigned short* p) {
    v8bf lo = *(const v8bf*)p;
    v8bf hi = *(const v8bf*)(p + 8);
    return __builtin_shufflevector(lo, hi, 0,1,2,3,4,5,6,7,8,9,10,11,12,13,14,15);
}
__device__ __forceinline__ v8f wmma_bf16(v16bf a, v16bf b, v8f c) {
    return __builtin_amdgcn_wmma_f32_16x16x32_bf16(false, a, false, b, (short)0, c,
                                                   false, false);
}

// ---- cross-lane reductions over the 16-lane C-fragment column groups ------
__device__ __forceinline__ float rowmax16(float v) {
    #pragma unroll
    for (int off = 1; off < 16; off <<= 1) v = fmaxf(v, __shfl_xor(v, off, 32));
    return v;
}
__device__ __forceinline__ float rowsum16(float v) {
    #pragma unroll
    for (int off = 1; off < 16; off <<= 1) v += __shfl_xor(v, off, 32);
    return v;
}

// ===========================================================================
// Kernel 1: f32 -> bf16 conversion (grid-strided)
// ===========================================================================
__global__ void k_cvt_bf16(const float* __restrict__ src,
                           unsigned short* __restrict__ dst, int n) {
    int i = blockIdx.x * blockDim.x + threadIdx.x;
    int stride = gridDim.x * blockDim.x;
    for (; i < n; i += stride) dst[i] = f2bf(src[i]);
}

// ===========================================================================
// Kernel 2/5: GEMM  out[m,o] = sum_k A[m,k] * W[o,k]  (+ bias, mode-specific
// epilogue).  A: [4096,1024] bf16 row-major.  W: [1024,1024] bf16 row-major
// (einsum 'bni,oi' => B-fragment column o reads W row o contiguously).
// Wave tile: 64(M) x 64(N), k-step 32 => 16 WMMAs / 16 b128 loads per iter.
// 8 waves/block, 1024 waves total per matrix.
// mode 0: q -> bf16 [B,H,N,D], scaled    mode 1: k -> bf16 [B,H,N,D]
// mode 2: v -> bf16 [B,H,D,N] (transposed for PV B-fragments)
// mode 3: final out -> f32 [TOK, DMODEL]
// ===========================================================================
__global__ __launch_bounds__(256) void k_gemm(
    const unsigned short* __restrict__ A,
    const unsigned short* __restrict__ w0, const unsigned short* __restrict__ w1,
    const unsigned short* __restrict__ w2,
    const float* __restrict__ b0, const float* __restrict__ b1,
    const float* __restrict__ b2,
    unsigned short* __restrict__ outq, unsigned short* __restrict__ outk,
    unsigned short* __restrict__ outvt, float* __restrict__ outf,
    float qscale, int is_final) {

    const int mode = is_final ? 3 : (int)blockIdx.z;
    const unsigned short* W   = (mode == 1) ? w1 : (mode == 2) ? w2 : w0;
    const float*          bia = (mode == 1) ? b1 : (mode == 2) ? b2 : b0;
    const float scale = (mode == 0) ? qscale : 1.0f;

    const int lane = threadIdx.x & 31;
    const int wv   = threadIdx.x >> 5;
    const int wid  = blockIdx.x * 8 + wv;            // 1024 waves total
    const int wN   = (wid & 15) * 64;                // 16 column strips
    const int wM   = (wid >> 4) * 64;                // 64 row strips
    const int row  = lane & 15;
    const int half = lane >> 4;
    const int koffA = half * 16;
    const int koffB = half * 16;

    v8f acc[4][4] = {};

    for (int k0 = 0; k0 < DMODEL; k0 += 32) {
        v16bf b[4];
        #pragma unroll
        for (int ct = 0; ct < 4; ct++)
            b[ct] = load_fragB(W + (size_t)(wN + ct * 16 + row) * DMODEL + k0 + koffB);
        #pragma unroll
        for (int rt = 0; rt < 4; rt++) {
            v16bf a = load_fragA(A + (size_t)(wM + rt * 16 + row) * DMODEL + k0 + koffA);
            #pragma unroll
            for (int ct = 0; ct < 4; ct++)
                acc[rt][ct] = wmma_bf16(a, b[ct], acc[rt][ct]);
        }
    }

    // epilogue: C layout -> reg r holds row (r + 8*half), column = lane&15
    #pragma unroll
    for (int ct = 0; ct < 4; ct++) {
        const int o = wN + ct * 16 + row;
        const float bsv = bia[o];
        #pragma unroll
        for (int rt = 0; rt < 4; rt++) {
            #pragma unroll
            for (int r = 0; r < 8; r++) {
                const int m = wM + rt * 16 + r + 8 * half;
                const float val = (acc[rt][ct][r] + bsv) * scale;
                if (mode == 3) {
                    outf[(size_t)m * DMODEL + o] = val;
                } else {
                    const int bb = m >> 11, np = m & (SEQ - 1);
                    const int hh = o >> 6,  dd = o & 63;
                    if (mode == 2)
                        outvt[(((size_t)(bb * NHEADS + hh)) * HDIM + dd) * SEQ + np] = f2bf(val);
                    else {
                        unsigned short* dst = (mode == 0) ? outq : outk;
                        dst[(((size_t)(bb * NHEADS + hh)) * SEQ + np) * HDIM + dd] = f2bf(val);
                    }
                }
            }
        }
    }
}

// ===========================================================================
// Kernel 3: interleaved RoPE in-place on q and k  ([B,H,N,D] bf16)
// out[2i] = t[2i]*cos - t[2i+1]*sin ; out[2i+1] = t[2i+1]*cos + t[2i]*sin
// ===========================================================================
__global__ void k_rope(unsigned short* __restrict__ q,
                       unsigned short* __restrict__ k,
                       const float* __restrict__ freqs) {
    const int total = BATCH * NHEADS * SEQ * (HDIM / 2);   // 2^21 pairs/tensor
    int p = blockIdx.x * blockDim.x + threadIdx.x;
    unsigned short* t = q;
    if (p >= total) { t = k; p -= total; }
    if (p >= total) return;
    const int i  = p & 31;               // freq index
    const int np = (p >> 5) & (SEQ - 1); // position
    const int bh = p >> 16;              // 0..31
    const size_t base = ((size_t)bh * SEQ + np) * HDIM + 2 * i;
    const float ang = (float)np * freqs[i];
    const float c = cosf(ang), s = sinf(ang);
    const float x0 = bf2f(t[base]), x1 = bf2f(t[base + 1]);
    t[base]     = f2bf(x0 * c - x1 * s);
    t[base + 1] = f2bf(x1 * c + x0 * s);
}

// ===========================================================================
// Kernel 4: flash attention per (b,h).  Block = 8 waves, each wave owns a
// 16-query tile (block covers 128 queries).  Keys streamed in chunks of 32.
// S = Q Kt via 4 WMMAs; online softmax (C-fragment reg r <-> row r, column
// reductions via shfl_xor over 16-lane groups); P re-laid out via LDS into an
// A fragment; O += P V via 4 WMMAs on transposed-V B fragments.
// ===========================================================================
__global__ __launch_bounds__(256) void k_attn(
    const unsigned short* __restrict__ Q,   // [B,H,N,D] bf16 (pre-scaled)
    const unsigned short* __restrict__ K,   // [B,H,N,D] bf16
    const unsigned short* __restrict__ VT,  // [B,H,D,N] bf16
    const unsigned char*  __restrict__ mask,// [B,N] bool
    unsigned short* __restrict__ attn) {    // [B,N,DMODEL] bf16 (heads merged)

    __shared__ __align__(16) unsigned short Pls[8][16 * 32];

    const int lane = threadIdx.x & 31;
    const int wv   = threadIdx.x >> 5;
    const int bh   = blockIdx.y;
    const int b    = bh >> 4;
    const int h    = bh & 15;
    const int qm   = blockIdx.x * 128 + wv * 16;     // query base in sequence
    const int row  = lane & 15;
    const int half = lane >> 4;
    const int koffA = half * 16;
    const int koffB = half * 16;

    const unsigned short* Qb = Q  + (size_t)bh * SEQ * HDIM;
    const unsigned short* Kb = K  + (size_t)bh * SEQ * HDIM;
    const unsigned short* Vb = VT + (size_t)bh * HDIM * SEQ;
    const unsigned char*  mb = mask + (size_t)b * SEQ;

    // Q A-fragments: rows qm..qm+15, K-dim = head dim (64 = 2 chunks of 32)
    const v16bf aq0 = load_fragA(Qb + (size_t)(qm + row) * HDIM + 0  + koffA);
    const v16bf aq1 = load_fragA(Qb + (size_t)(qm + row) * HDIM + 32 + koffA);

    float mi[8], li[8];
    v8f Oa[4] = {};
    #pragma unroll
    for (int r = 0; r < 8; r++) { mi[r] = -1e30f; li[r] = 0.0f; }

    for (int kb = 0; kb < SEQ; kb += 32) {
        // ---- scores for 32 keys: two 16x16 C tiles --------------------------
        v8f S0 = {}, S1 = {};
        {
            v16bf b0lo = load_fragB(Kb + (size_t)(kb + row) * HDIM + 0  + koffB);
            v16bf b0hi = load_fragB(Kb + (size_t)(kb + row) * HDIM + 32 + koffB);
            S0 = wmma_bf16(aq0, b0lo, S0);
            S0 = wmma_bf16(aq1, b0hi, S0);
            v16bf b1lo = load_fragB(Kb + (size_t)(kb + 16 + row) * HDIM + 0  + koffB);
            v16bf b1hi = load_fragB(Kb + (size_t)(kb + 16 + row) * HDIM + 32 + koffB);
            S1 = wmma_bf16(aq0, b1lo, S1);
            S1 = wmma_bf16(aq1, b1hi, S1);
        }
        // ---- padding mask (column = lane&15 within each tile) ---------------
        const bool msk0 = mb[kb + row] != 0;
        const bool msk1 = mb[kb + 16 + row] != 0;
        #pragma unroll
        for (int r = 0; r < 8; r++) {
            S0[r] = msk0 ? -1e30f : S0[r];
            S1[r] = msk1 ? -1e30f : S1[r];
        }
        // ---- online softmax update ------------------------------------------
        float P0[8], P1[8];
        #pragma unroll
        for (int r = 0; r < 8; r++) {
            const float rmax = rowmax16(fmaxf(S0[r], S1[r]));
            const float mnew = fmaxf(mi[r], rmax);
            const float alpha = __expf(mi[r] - mnew);
            mi[r] = mnew;
            const float p0 = __expf(S0[r] - mnew);
            const float p1 = __expf(S1[r] - mnew);
            P0[r] = p0; P1[r] = p1;
            const float rs = rowsum16(p0 + p1);
            li[r] = li[r] * alpha + rs;
            #pragma unroll
            for (int dt = 0; dt < 4; dt++) Oa[dt][r] *= alpha;
        }
        // ---- P: C fragment -> LDS -> A fragment -----------------------------
        __syncthreads();
        #pragma unroll
        for (int r = 0; r < 8; r++) {
            const int ml = r + 8 * half;
            Pls[wv][ml * 32 + row]      = f2bf(P0[r]);
            Pls[wv][ml * 32 + 16 + row] = f2bf(P1[r]);
        }
        __syncthreads();
        const v16bf ap = load_fragA(&Pls[wv][row * 32 + koffA]);
        // ---- O += P * V  (V transposed: B-fragment columns = head dims) -----
        #pragma unroll
        for (int dt = 0; dt < 4; dt++) {
            v16bf bv = load_fragB(Vb + (size_t)(dt * 16 + row) * SEQ + kb + koffB);
            Oa[dt] = wmma_bf16(ap, bv, Oa[dt]);
        }
    }

    // ---- normalize + store merged-head output ------------------------------
    #pragma unroll
    for (int dt = 0; dt < 4; dt++) {
        #pragma unroll
        for (int r = 0; r < 8; r++) {
            const int m  = qm + r + 8 * half;
            const int dd = dt * 16 + row;
            const float val = Oa[dt][r] / li[r];
            attn[((size_t)(b * SEQ + m)) * DMODEL + h * HDIM + dd] = f2bf(val);
        }
    }
}

// ===========================================================================
// Host-side launch
// ===========================================================================
extern "C" void kernel_launch(void* const* d_in, const int* in_sizes, int n_in,
                              void* d_out, int out_size, void* d_ws, size_t ws_size,
                              hipStream_t stream) {
    const float*         x     = (const float*)d_in[0];
    const unsigned char* pmask = (const unsigned char*)d_in[1];
    const float* Wq = (const float*)d_in[2];
    const float* bq = (const float*)d_in[3];
    const float* Wk = (const float*)d_in[4];
    const float* bk = (const float*)d_in[5];
    const float* Wv = (const float*)d_in[6];
    const float* bv = (const float*)d_in[7];
    const float* Wo = (const float*)d_in[8];
    const float* bo = (const float*)d_in[9];
    const float* fr = (const float*)d_in[10];
    float* out = (float*)d_out;

    // workspace layout (bf16 elements)
    unsigned short* ws = (unsigned short*)d_ws;
    const size_t XB = 0;                               // x bf16      [4096,1024]
    const size_t WQ = XB + (size_t)TOK * DMODEL;       // Wq bf16     [1024,1024]
    const size_t WK = WQ + (size_t)DMODEL * DMODEL;
    const size_t WV = WK + (size_t)DMODEL * DMODEL;
    const size_t WO = WV + (size_t)DMODEL * DMODEL;
    const size_t QH = WO + (size_t)DMODEL * DMODEL;    // q [B,H,N,D]
    const size_t KH = QH + (size_t)TOK * DMODEL;       // k [B,H,N,D]
    const size_t VT = KH + (size_t)TOK * DMODEL;       // v [B,H,D,N]
    const size_t AT = VT + (size_t)TOK * DMODEL;       // attn out [B,N,DMODEL]

    // 1) convert inputs to bf16
    k_cvt_bf16<<<2048, 256, 0, stream>>>(x,  ws + XB, TOK * DMODEL);
    k_cvt_bf16<<<512,  256, 0, stream>>>(Wq, ws + WQ, DMODEL * DMODEL);
    k_cvt_bf16<<<512,  256, 0, stream>>>(Wk, ws + WK, DMODEL * DMODEL);
    k_cvt_bf16<<<512,  256, 0, stream>>>(Wv, ws + WV, DMODEL * DMODEL);
    k_cvt_bf16<<<512,  256, 0, stream>>>(Wo, ws + WO, DMODEL * DMODEL);

    // 2) fused QKV projections (z = 0:q, 1:k, 2:v), 1024 waves each
    k_gemm<<<dim3(128, 1, 3), 256, 0, stream>>>(
        ws + XB, ws + WQ, ws + WK, ws + WV, bq, bk, bv,
        ws + QH, ws + KH, ws + VT, nullptr, QSCALE, 0);

    // 3) RoPE on q and k
    k_rope<<<16384, 256, 0, stream>>>(ws + QH, ws + KH, fr);

    // 4) flash attention (grid: 16 query blocks x 32 (b,h) pairs)
    k_attn<<<dim3(16, 32), 256, 0, stream>>>(
        ws + QH, ws + KH, ws + VT, pmask, ws + AT);

    // 5) output projection -> f32
    k_gemm<<<dim3(128, 1, 1), 256, 0, stream>>>(
        ws + AT, ws + WO, nullptr, nullptr, bo, nullptr, nullptr,
        nullptr, nullptr, nullptr, out, 1.0f, 1);
}
